// attention_bayes_54150947668556
// MI455X (gfx1250) — matrix operand order repaired and checked by reading, not compiled, not run on hardware
//
#include <hip/hip_runtime.h>
#include <hip/hip_bf16.h>

typedef __attribute__((ext_vector_type(16))) _Float16 v16h;
typedef __attribute__((ext_vector_type(8)))  float    v8f;
typedef unsigned int u32x4 __attribute__((ext_vector_type(4)));
typedef int          i32x4 __attribute__((ext_vector_type(4)));
typedef int          i32x8 __attribute__((ext_vector_type(8)));

static constexpr int BB  = 8;
static constexpr int NQ  = 2048;
static constexpr int NK2 = 2048;
static constexpr int DM  = 64;

// ---------------- feature detection (compile-safe fallbacks) ----------------
#ifndef __has_builtin
#define __has_builtin(x) 0
#endif

#if __has_builtin(__builtin_amdgcn_tensor_load_to_lds) && \
    __has_builtin(__builtin_amdgcn_s_wait_tensorcnt)
#define USE_TDM 1
#else
#define USE_TDM 0
#endif

#if __has_builtin(__builtin_amdgcn_global_load_async_to_lds_b128) && \
    __has_builtin(__builtin_amdgcn_s_wait_asynccnt)
#define USE_ASYNC_LD 1
#else
#define USE_ASYNC_LD 0
#endif

#if __has_builtin(__builtin_amdgcn_global_store_async_from_lds_b128) && \
    __has_builtin(__builtin_amdgcn_s_wait_asynccnt)
#define USE_ASYNC_ST 1
#else
#define USE_ASYNC_ST 0
#endif

// async builtins take int4 pointers: (v4i AS1*, v4i AS3*, imm, imm)
typedef __attribute__((address_space(1))) i32x4 gi4;
typedef __attribute__((address_space(3))) i32x4 li4;

__device__ __forceinline__ gi4* to_g4(const void* p) {
    return (gi4*)(unsigned long long)p;
}
__device__ __forceinline__ li4* to_l4(void* p) {
    // AS3 pointers are 32-bit; low 32 bits of a flat LDS pointer are the DS address
    return (li4*)(unsigned)(unsigned long long)p;
}

// ---------- WMMA fragment helpers (CDNA5 16x16x32 f16 layouts, ISA 7.12.2) ----------

__device__ __forceinline__ int a_kidx(int p, int h) {
    return h * 16 + 2 * p;
}

__device__ __forceinline__ v16h a_from_f32(const float* row, int k0, int h) {
    v16h a;
#pragma unroll
    for (int p = 0; p < 8; ++p) {
        int K = k0 + a_kidx(p, h);
        a[2 * p]     = (_Float16)row[K];
        a[2 * p + 1] = (_Float16)row[K + 1];
    }
    return a;
}

__device__ __forceinline__ v16h a_from_f16(const _Float16* row, int k0, int h) {
    v16h a;
#pragma unroll
    for (int p = 0; p < 8; ++p) {
        int K = k0 + a_kidx(p, h);
        a[2 * p]     = row[K];
        a[2 * p + 1] = row[K + 1];
    }
    return a;
}

// B fragment: lane holds column N=l&15, K-rows (l>>4)*16..+15; with K-major
// storage the 16 halves are contiguous -> b128 loads.
__device__ __forceinline__ v16h b_contig_f16(const _Float16* p0) {
    v16h b;
#pragma unroll
    for (int i = 0; i < 16; ++i) b[i] = p0[i];
    return b;
}

__device__ __forceinline__ v8f wmma_f16(v16h a, v16h b, v8f c) {
    return __builtin_amdgcn_wmma_f32_16x16x32_f16(false, a, false, b, (short)0, c, false, false);
}

// ---------------- math helpers ----------------

__device__ __forceinline__ float softplusf(float x) {
    return (x > 20.f) ? x : log1pf(__expf(x));
}

// digamma for x >= 1 (alpha = qk+1 > 1; q,k are softplus outputs)
__device__ __forceinline__ float digammaf_pos(float x) {
    float s = 0.f;
    while (x < 6.f) { s -= 1.f / x; x += 1.f; }
    float inv = 1.f / x, i2 = inv * inv;
    return s + __logf(x) - 0.5f * inv
           - i2 * (1.f / 12.f - i2 * (1.f / 120.f - i2 * (1.f / 252.f)));
}

// ---------------- Stage 0: transpose weights to f16 K-major ----------------
// Wt[which][l][n][d] so every B fragment is 16 contiguous halves.

__global__ void wprep_kernel(const float* __restrict__ Wq,
                             const float* __restrict__ Wk,
                             const float* __restrict__ Wv,
                             _Float16* __restrict__ Wt) {
    const int which = blockIdx.x, l = blockIdx.y;
    const float* W = ((which == 0) ? Wq : (which == 1) ? Wk : Wv) + l * DM * DM;
    _Float16* T = Wt + (size_t)(which * 3 + l) * DM * DM;
    for (int i = threadIdx.x; i < DM * DM; i += 256) {
        int d = i >> 6, n = i & 63;
        T[n * DM + d] = (_Float16)W[d * DM + n];
    }
}

// ---------------- Stage 1: three 3-layer MLPs (q from y, k/va from x) ----------------

__global__ void __launch_bounds__(256) mlp3_kernel(
    const float* __restrict__ y,  const float* __restrict__ x,
    const _Float16* __restrict__ Wt,
    const float* __restrict__ bq, const float* __restrict__ bk,
    const float* __restrict__ bv,
    _Float16* __restrict__ qh, _Float16* __restrict__ kh,
    _Float16* __restrict__ vhT, float* __restrict__ va_f32)
{
    __shared__ float scratch[8][16 * DM];   // 32 KB C->A transpose staging

    const int tid = threadIdx.x, wave = tid >> 5, lane = tid & 31;
    const int l15 = lane & 15, h = lane >> 4;
    const int which = blockIdx.y;           // 0:q(y) 1:k(x) 2:va(x)

    const float* in  = (which == 0) ? y  : x;
    const float* bia = (which == 0) ? bq : (which == 1) ? bk : bv;
    _Float16*    oh  = (which == 0) ? qh : (which == 1) ? kh : (_Float16*)nullptr;
    const bool last_sp = (which != 2);

    const int row0 = blockIdx.x * 128 + wave * 16;

    const float* arow = in + (size_t)(row0 + l15) * DM;
    v16h a0 = a_from_f32(arow, 0, h);
    v16h a1 = a_from_f32(arow, 32, h);

    for (int l = 0; l < 3; ++l) {
        const _Float16* Wl = Wt + (size_t)(which * 3 + l) * DM * DM;  // [n][d]
        v8f c[4] = {};
#pragma unroll
        for (int nt = 0; nt < 4; ++nt) {
            const _Float16* bcol = Wl + (size_t)(nt * 16 + l15) * DM;
            v16h b0 = b_contig_f16(bcol + h * 16);          // K = 0..31
            v16h b1 = b_contig_f16(bcol + 32 + h * 16);     // K = 32..63
            c[nt] = wmma_f16(a0, b0, c[nt]);
            c[nt] = wmma_f16(a1, b1, c[nt]);
        }
        const bool act = (l < 2) || last_sp;
#pragma unroll
        for (int nt = 0; nt < 4; ++nt) {
            float bs = bia[l * DM + nt * 16 + l15];
#pragma unroll
            for (int v = 0; v < 8; ++v) {
                float val = c[nt][v] + bs;
                if (act) val = softplusf(val);
                scratch[wave][(v + h * 8) * DM + nt * 16 + l15] = val;
            }
        }
        __syncthreads();
        if (l < 2) {
            const float* srow = &scratch[wave][l15 * DM];
            a0 = a_from_f32(srow, 0, h);
            a1 = a_from_f32(srow, 32, h);
        }
        __syncthreads();
    }

    for (int i = lane; i < 16 * DM; i += 32) {
        int r = i >> 6, cc = i & 63;
        float val = scratch[wave][i];
        int grow = row0 + r;
        if (which == 2) {
            va_f32[(size_t)grow * DM + cc] = val;
            int b2 = grow >> 11, lr = grow & 2047;          // vhT: [B][DM][NK]
            vhT[(size_t)b2 * DM * NK2 + (size_t)cc * NK2 + lr] = (_Float16)val;
        } else {
            oh[(size_t)grow * DM + cc] = (_Float16)val;
        }
    }
}

// ---------------- Stage 2: fused scores / Dirichlet / softmax / attn@va ----------------
// LDS: A[16*2048] f32 (alpha->z->attn), Mf[16*2048] f32 mask (aliased by GEMM
// partials P[8*1024] in phase F), ST[16*8] stats. Total 262656 B < 320 KB/WGP.

__global__ void __launch_bounds__(256) attn_kernel(
    const float* __restrict__ mask, const float* __restrict__ eps,
    const _Float16* __restrict__ qh, const _Float16* __restrict__ kh,
    const _Float16* __restrict__ vhT,
    float* __restrict__ out, float* __restrict__ attn,
    float* __restrict__ kldpart)
{
    extern __shared__ float lds[];
    float* A  = lds;                    // 32768 floats
    float* Mf = lds + 16 * NK2;         // 32768 floats (mask)
    float* P  = Mf;                     // phase-F partials alias dead mask
    float* ST = lds + 2 * 16 * NK2;     // 16*8 floats

    const int tid = threadIdx.x, wave = tid >> 5, lane = tid & 31;
    const int l15 = lane & 15, h = lane >> 4;
    const int b = blockIdx.x >> 7, qt = blockIdx.x & 127;
    const int q0 = qt * 16;
    const size_t rowbase = (size_t)(b * NQ + q0);

    const float* mrow = mask + rowbase * NK2;

    // ---- mask tile -> LDS via TDM / async-DMA / fallback ----
#if USE_TDM
    if (wave == 0) {
        unsigned ldsa = (unsigned)(unsigned long long)Mf;
        unsigned long long ga = (unsigned long long)mrow;
        u32x4 g0;
        g0[0] = 1u;                                            // count=1
        g0[1] = ldsa;                                          // lds_addr
        g0[2] = (unsigned)(ga & 0xFFFFFFFFull);                // global_addr lo
        g0[3] = (unsigned)((ga >> 32) & 0x01FFFFFFull) | (2u << 30);  // hi | type=2
        i32x8 g1;
        g1[0] = (int)(2u << 16);                   // data_size = 4 B
        g1[1] = (int)((2048u & 0xFFFFu) << 16);    // tensor_dim0 lo16
        g1[2] = (int)((2048u >> 16) | (16u << 16));// dim0 hi16 | tensor_dim1 lo16
        g1[3] = (int)(2048u << 16);                // dim1 hi16 | tile_dim0=2048
        g1[4] = (int)16u;                          // tile_dim1=16, tile_dim2=0
        g1[5] = (int)2048u;                        // tensor_dim0_stride lo32
        g1[6] = 0;
        g1[7] = 0;
        i32x4 z4 = {0, 0, 0, 0};
#if __clang_major__ >= 23
        i32x8 z8 = {0, 0, 0, 0, 0, 0, 0, 0};
        __builtin_amdgcn_tensor_load_to_lds(g0, g1, z4, z4, z8, 0);
#else
        __builtin_amdgcn_tensor_load_to_lds(g0, g1, z4, z4, 0);
#endif
    }
#elif USE_ASYNC_LD
    for (int j = tid; j < (16 * NK2) / 4; j += 256)
        __builtin_amdgcn_global_load_async_to_lds_b128(
            to_g4(mrow + j * 4), to_l4(Mf + j * 4), 0, 0);
#else
    for (int i = tid; i < 16 * NK2; i += 256) Mf[i] = mrow[i];
#endif

    // prefetch the eps tile (consumed in phase C) while WMMA runs
    {
        const float* erow0 = eps + rowbase * NK2;
        for (int j = tid; j < (16 * NK2) / 32; j += 256)
            __builtin_prefetch(erow0 + j * 32, 0, 3);
    }

    // ---- phase A: alpha = q.k^T + 1 (WMMA), each wave owns 256 columns ----
    const _Float16* qrow = qh + (rowbase + l15) * DM;
    v16h a0 = a_from_f16(qrow, 0, h);
    v16h a1 = a_from_f16(qrow, 32, h);
    const _Float16* kb = kh + (size_t)b * NK2 * DM;
    for (int nt = 0; nt < 16; ++nt) {
        int n0 = wave * 256 + nt * 16;
        const _Float16* krow = kb + (size_t)(n0 + l15) * DM;   // k[n][d]: K contiguous
        v16h b0 = b_contig_f16(krow + h * 16);
        v16h b1 = b_contig_f16(krow + 32 + h * 16);
        v8f c = {};
        c = wmma_f16(a0, b0, c);
        c = wmma_f16(a1, b1, c);
#pragma unroll
        for (int v = 0; v < 8; ++v)
            A[(v + h * 8) * NK2 + n0 + l15] = c[v] + 1.0f;
    }
#if USE_TDM
    if (wave == 0) __builtin_amdgcn_s_wait_tensorcnt(0);
#elif USE_ASYNC_LD
    __builtin_amdgcn_s_wait_asynccnt(0);
#endif
    __syncthreads();

    // ---- phase B: per-row Dirichlet sums (16 threads per row) ----
    const int r = tid >> 4, t16 = tid & 15;
    float S_a = 0, S_lg = 0, S_adg = 0, S_dg = 0, S_m = 0, S_log = 0, S_inv = 0;
    for (int cc = t16; cc < NK2; cc += 16) {
        float a  = A[r * NK2 + cc];
        float m  = Mf[r * NK2 + cc];
        float la = __logf(a);
        float ia = 1.f / a;
        float dg = digammaf_pos(a);
        float lg = lgammaf(a);
        S_a += a * m;  S_lg += lg * m;  S_adg += a * dg * m;
        S_dg += dg * m; S_m += m; S_log += la * m; S_inv += ia * m;
    }
#pragma unroll
    for (int off = 8; off >= 1; off >>= 1) {
        S_a   += __shfl_xor(S_a, off);   S_lg  += __shfl_xor(S_lg, off);
        S_adg += __shfl_xor(S_adg, off); S_dg  += __shfl_xor(S_dg, off);
        S_m   += __shfl_xor(S_m, off);   S_log += __shfl_xor(S_log, off);
        S_inv += __shfl_xor(S_inv, off);
    }
    if (t16 == 0) {
        float a0s = S_a, latent = S_m;
        float dg0 = digammaf_pos(a0s);
        float kl = lgammaf(a0s) - S_lg + S_adg - dg0 * a0s
                 - (S_dg - dg0 * latent) * (1.f / (float)NK2);
        ST[r * 8 + 0] = S_log * (1.f / (float)NK2);
        ST[r * 8 + 1] = 1.f - 2.f / latent;
        ST[r * 8 + 2] = S_inv / (latent * latent);
        ST[r * 8 + 5] = fabsf(kl);
    }
    __syncthreads();
    if (tid == 0) {                     // deterministic per-block kld partial
        float s = 0;
        for (int i = 0; i < 16; ++i) s += ST[i * 8 + 5];
        kldpart[blockIdx.x] = s * (1.f / (float)(BB * NQ));
    }

    // ---- phase C: z = mu + sigma*eps + neg, row max ----
    const float mlog = ST[r * 8 + 0], c1 = ST[r * 8 + 1], c2 = ST[r * 8 + 2];
    const float* erow = eps + (rowbase + r) * NK2;
    float mx = -3.4e38f;
    for (int cc = t16; cc < NK2; cc += 16) {
        float a  = A[r * NK2 + cc];
        float m  = Mf[r * NK2 + cc];
        float la = __logf(a);
        float z  = (la - mlog) * m + (c1 / a + c2) * m * erow[cc]
                 + (1.f - m) * (-1e9f);
        A[r * NK2 + cc] = z;
        mx = fmaxf(mx, z);
    }
#pragma unroll
    for (int off = 8; off >= 1; off >>= 1) mx = fmaxf(mx, __shfl_xor(mx, off));
    if (t16 == 0) ST[r * 8 + 3] = mx;
    __syncthreads();

    // ---- phase D: exp + row sum ----
    mx = ST[r * 8 + 3];
    float se = 0.f;
    for (int cc = t16; cc < NK2; cc += 16) {
        float e = __expf(A[r * NK2 + cc] - mx);
        A[r * NK2 + cc] = e;
        se += e;
    }
#pragma unroll
    for (int off = 8; off >= 1; off >>= 1) se += __shfl_xor(se, off);
    if (t16 == 0) ST[r * 8 + 4] = se;
    __syncthreads();

    // ---- phase E: normalize in place, stream attn out ----
    {
        float invs = 1.f / ST[r * 8 + 4];
#if USE_ASYNC_ST
        for (int cc = t16; cc < NK2; cc += 16) A[r * NK2 + cc] *= invs;
#else
        float* arow = attn + (rowbase + r) * NK2;
        for (int cc = t16; cc < NK2; cc += 16) {
            float v = A[r * NK2 + cc] * invs;
            A[r * NK2 + cc] = v;
            arow[cc] = v;
        }
#endif
    }
    __syncthreads();
#if USE_ASYNC_ST
    {
        float* ag = attn + rowbase * NK2;   // contiguous 128 KB tile
        for (int j = tid; j < (16 * NK2) / 4; j += 256)
            __builtin_amdgcn_global_store_async_from_lds_b128(
                to_g4(ag + j * 4), to_l4(A + j * 4), 0, 0);
    }
#endif

    // ---- phase F: out = attn @ va (normalized already), K split across waves ----
    v8f acc[4] = {};
    const _Float16* vbT = vhT + (size_t)b * DM * NK2;
    const int kw = wave * 256;
    for (int ks = 0; ks < 8; ++ks) {
        int k0 = kw + ks * 32;
        v16h af = a_from_f32(A + l15 * NK2 + k0, 0, h);
#pragma unroll
        for (int nt = 0; nt < 4; ++nt) {
            const _Float16* bcol = vbT + (size_t)(nt * 16 + l15) * NK2 + k0;
            acc[nt] = wmma_f16(af, b_contig_f16(bcol + h * 16), acc[nt]);
        }
    }
#pragma unroll
    for (int nt = 0; nt < 4; ++nt)
#pragma unroll
        for (int v = 0; v < 8; ++v)
            P[wave * 1024 + (v + h * 8) * DM + nt * 16 + l15] = acc[nt][v];
    __syncthreads();
    for (int o = tid * 4; o < tid * 4 + 4; ++o) {
        float s = 0;
#pragma unroll
        for (int w2 = 0; w2 < 8; ++w2) s += P[w2 * 1024 + o];
        int rr = o >> 6, cc = o & 63;
        out[(rowbase + rr) * DM + cc] = s;
    }
#if USE_ASYNC_ST
    __builtin_amdgcn_s_wait_asynccnt(0);
#endif
}

// ---------------- Stage 3: deterministic kld reduction ----------------

__global__ void kld_reduce_kernel(const float* __restrict__ part,
                                  float* __restrict__ kld_out, int n) {
    if (threadIdx.x == 0 && blockIdx.x == 0) {
        float s = 0.f;
        for (int i = 0; i < n; ++i) s += part[i];
        *kld_out = s;
    }
}

// ---------------- launch ----------------

extern "C" void kernel_launch(void* const* d_in, const int* in_sizes, int n_in,
                              void* d_out, int out_size, void* d_ws, size_t ws_size,
                              hipStream_t stream) {
    const float* y    = (const float*)d_in[0];
    const float* x    = (const float*)d_in[1];
    const float* mask = (const float*)d_in[2];
    const float* eps  = (const float*)d_in[3];
    const float* Wq   = (const float*)d_in[4];
    const float* bq   = (const float*)d_in[5];
    const float* Wk   = (const float*)d_in[6];
    const float* bk   = (const float*)d_in[7];
    const float* Wv   = (const float*)d_in[8];
    const float* bv   = (const float*)d_in[9];

    float* out  = (float*)d_out;
    float* attn = out  + (size_t)BB * NQ * DM;
    float* va   = attn + (size_t)BB * NQ * NK2;
    float* kld  = va   + (size_t)BB * NK2 * DM;

    _Float16* qh  = (_Float16*)d_ws;                       // 2 MB
    _Float16* kh  = qh  + (size_t)BB * NQ * DM;            // 2 MB
    _Float16* vhT = kh  + (size_t)BB * NK2 * DM;           // 2 MB ([B][DM][NK])
    _Float16* Wt  = vhT + (size_t)BB * NK2 * DM;           // 72 KB
    float* kldpart = (float*)(Wt + (size_t)9 * DM * DM);   // 4 KB

    wprep_kernel<<<dim3(3, 3), 256, 0, stream>>>(Wq, Wk, Wv, Wt);

    mlp3_kernel<<<dim3(128, 3), 256, 0, stream>>>(
        y, x, Wt, bq, bk, bv, qh, kh, vhT, va);

    size_t shmem = (size_t)(2 * 16 * NK2 + 16 * 8) * 4;    // 262656 B
    attn_kernel<<<dim3(BB * (NQ / 16)), 256, shmem, stream>>>(
        mask, eps, qh, kh, vhT, out, attn, kldpart);

    kld_reduce_kernel<<<1, 64, 0, stream>>>(kldpart, kld, BB * (NQ / 16));
}
